// GRUQNetwork_54116587929780
// MI455X (gfx1250) — matrix operand order repaired and checked by reading, not compiled, not run on hardware
//
#include <hip/hip_runtime.h>
#include <stdint.h>

#define T_STEPS 2048
#define OBS     512
#define MLPD    1024
#define HID     1024
#define ACT_DIM 16

typedef __attribute__((ext_vector_type(16))) _Float16     v16h;
typedef __attribute__((ext_vector_type(8)))  float        v8f;
typedef __attribute__((ext_vector_type(4)))  float        f32x4;
typedef __attribute__((ext_vector_type(4)))  unsigned int u32x4;
typedef __attribute__((ext_vector_type(4)))  int          i32x4;
typedef __attribute__((ext_vector_type(8)))  int          i32x8;

#ifndef __has_builtin
#define __has_builtin(x) 0
#endif
#if __has_builtin(__builtin_amdgcn_tensor_load_to_lds) && \
    __has_builtin(__builtin_amdgcn_s_wait_tensorcnt)
#define HAVE_TDM 1
#else
#define HAVE_TDM 0
#endif

// ---------------- TDM helpers (Tensor Data Mover, gfx1250) ------------------
#if HAVE_TDM
__device__ __forceinline__ uint32_t lds_off(const void* p) {
    // generic pointer into LDS: low 32 bits == LDS byte address (flat aperture)
    return (uint32_t)(uintptr_t)p;
}
// 2D tile load: tile_d0 elements per row, tile_d1 rows, row stride (elements).
// elt_log2: 1 = f16, 2 = f32. D# per CDNA5 ISA ch.8 (groups 2/3 NULL => <=2D).
__device__ __forceinline__ void tdm_load_2d(uint32_t lds_byte_off, const void* gptr,
                                            uint32_t elt_log2, uint32_t tile_d0,
                                            uint32_t tile_d1, uint32_t stride0) {
    uint64_t ga = (uint64_t)(uintptr_t)gptr;
    u32x4 g0;
    g0[0] = 1u;                                            // count=1, user desc
    g0[1] = lds_byte_off;                                  // lds_addr
    g0[2] = (uint32_t)ga;                                  // global_addr[31:0]
    g0[3] = (uint32_t)((ga >> 32) & 0x01ffffffu) | (2u << 30); // addr[56:32]|type=2
    i32x8 g1;
    g1[0] = (int)(elt_log2 << 16);                         // wg_mask=0, data_size
    g1[1] = (int)((tile_d0 & 0xffffu) << 16);              // tensor_dim0[15:0]
    g1[2] = (int)(((tile_d0 >> 16) & 0xffffu) |            // tensor_dim0[31:16]
                  ((tile_d1 & 0xffffu) << 16));            // tensor_dim1[15:0]
    g1[3] = (int)(((tile_d1 >> 16) & 0xffffu) |            // tensor_dim1[31:16]
                  ((tile_d0 & 0xffffu) << 16));            // tile_dim0
    g1[4] = (int)(tile_d1 & 0xffffu);                      // tile_dim1 (tile_dim2=0)
    g1[5] = (int)stride0;                                  // tensor_dim0_stride lo
    g1[6] = 0;                                             // stride hi / dim1_stride
    g1[7] = 0;
    i32x4 z4 = {0, 0, 0, 0};
#if __clang_major__ >= 23
    i32x8 z8 = {};
    __builtin_amdgcn_tensor_load_to_lds(g0, g1, z4, z4, z8, 0);
#else
    __builtin_amdgcn_tensor_load_to_lds(g0, g1, z4, z4, 0);
#endif
}
__device__ __forceinline__ void tdm_wait() {
    __builtin_amdgcn_s_wait_tensorcnt(0);                  // s_wait_tensorcnt 0
}
#endif

// ---------------- deterministic pseudo-gaussian (NoisyLinear noise) ----------
__device__ __forceinline__ uint32_t hash32(uint32_t x) {
    x ^= x >> 17; x *= 0xed5ad4bbu;
    x ^= x >> 11; x *= 0xac4c1b51u;
    x ^= x >> 15; x *= 0x31848babu;
    x ^= x >> 14; return x;
}
__device__ __forceinline__ float gaussian(uint32_t seed, uint32_t idx) {
    uint32_t a = hash32(seed ^ (idx * 2u + 0x68bc21ebu));
    uint32_t b = hash32(seed ^ (idx * 2u + 0x2c1b3c6du));
    float u1 = ((a >> 8) + 1u) * (1.0f / 16777216.0f);
    float u2 = (b >> 8) * (1.0f / 16777216.0f);
    return sqrtf(-2.0f * __logf(u1)) * __cosf(6.28318530718f * u2);
}
__device__ __forceinline__ float mishf(float x) {
    float sp = __logf(1.0f + __expf(x));   // softplus (inf-safe: tanh(inf)=1)
    return x * tanhf(sp);
}

// ---------------- noisy-weight materialization / f16 conversion -------------
__global__ void k_noisy_f16(const float* __restrict__ wm, const float* __restrict__ ws,
                            _Float16* __restrict__ out, int n, uint32_t seed) {
    int i = blockIdx.x * blockDim.x + threadIdx.x;
    if (i >= n) return;
    float v = wm[i];
    if (ws) v += ws[i] * gaussian(seed, (uint32_t)i);
    out[i] = (_Float16)v;
}
__global__ void k_noisy_f32(const float* __restrict__ wm, const float* __restrict__ ws,
                            float* __restrict__ out, int n, uint32_t seed) {
    int i = blockIdx.x * blockDim.x + threadIdx.x;
    if (i >= n) return;
    float v = wm[i];
    if (ws) v += ws[i] * gaussian(seed, (uint32_t)i);
    out[i] = v;
}

// ---------------- WMMA GEMM: C[M,N] = act(A[M,K] @ B[N,K]^T + bias[N]) ------
// 8 waves/block share one 16-column B panel: staged into LDS (TDM double-
// buffered when available), fragments read back with ds_load_b128. Each wave
// owns one 16x16 C tile; f32 accumulate via v_wmma_f32_16x16x32_f16.
template <int DO_MISH>
__global__ void __launch_bounds__(256)
k_wmma_gemm(const _Float16* __restrict__ A, const _Float16* __restrict__ B,
            const float* __restrict__ bias,
            _Float16* __restrict__ Ch, float* __restrict__ Cf,
            int M, int N, int K) {
    constexpr int KC = 64;                       // K-chunk staged per block
    __shared__ __align__(16) _Float16 Blds[2][16 * KC];
    const int tid  = threadIdx.x;
    const int lane = tid & 31;
    const int wave = tid >> 5;
    const int g = lane >> 4;                     // half-wave group
    const int r = lane & 15;
    const int n0 = blockIdx.x * 16;
    const int m0 = (blockIdx.y * 8 + wave) * 16; // grids are exact: m0 < M

    const _Float16* Arow  = A + (size_t)(m0 + r) * K;  // A: row m0+r (global)
    const _Float16* Btile = B + (size_t)n0 * K;        // 16 weight rows

    // ---- stage chunk 0 ----
#if HAVE_TDM
    if (wave == 0) {
        tdm_load_2d(lds_off(&Blds[0][0]), Btile, 1, KC, 16, (uint32_t)K);
        tdm_wait();
    }
#else
    for (int idx = tid; idx < 16 * KC / 8; idx += 256) {   // 128 x b128
        int row = idx >> 3, qi = idx & 7;
        ((f32x4*)&Blds[0][0])[idx] = *(const f32x4*)(Btile + (size_t)row * K + qi * 8);
    }
#endif
    __syncthreads();

    v8f acc = {};
    const int nchunks = K / KC;
    for (int c = 0; c < nchunks; ++c) {
        const int buf = c & 1, nbuf = buf ^ 1;
        // ---- kick off next chunk while computing this one ----
        if (c + 1 < nchunks) {
#if HAVE_TDM
            if (wave == 0)
                tdm_load_2d(lds_off(&Blds[nbuf][0]), Btile + (c + 1) * KC,
                            1, KC, 16, (uint32_t)K);
#else
            for (int idx = tid; idx < 16 * KC / 8; idx += 256) {
                int row = idx >> 3, qi = idx & 7;
                ((f32x4*)&Blds[nbuf][0])[idx] =
                    *(const f32x4*)(Btile + (size_t)row * K + (c + 1) * KC + qi * 8);
            }
#endif
            __builtin_prefetch(Arow + (c + 1) * KC, 0, 1);     // global_prefetch_b8
        }
        // ---- compute: 2 WMMA k-tiles out of the staged chunk ----
        const int kbase = c * KC;
        const _Float16* Bl = &Blds[buf][r * KC + g * 8];       // ds_load source
#pragma unroll
        for (int kk = 0; kk < KC; kk += 32) {
            union { v16h v; f32x4 q[2]; } ua, ub;
            // A lane layout: halves 0-7 -> K+g*8.., halves 8-15 -> K+16+g*8..
            ua.q[0] = *(const f32x4*)(Arow + kbase + kk + g * 8);
            ua.q[1] = *(const f32x4*)(Arow + kbase + kk + 16 + g * 8);
            // B lane layout: halves 0-7 -> kk+g*8.., halves 8-15 -> kk+16+g*8.. (from LDS)
            ub.q[0] = *(const f32x4*)(Bl + kk);
            ub.q[1] = *(const f32x4*)(Bl + kk + 16);
            acc = __builtin_amdgcn_wmma_f32_16x16x32_f16(
                false, ua.v, false, ub.v, (short)0, acc, false, false);
        }
        __syncthreads();                          // done reading buf / stores visible
#if HAVE_TDM
        if (c + 1 < nchunks) {
            if (wave == 0) tdm_wait();            // nbuf landed in LDS
            __syncthreads();                      // publish to all waves
        }
#endif
    }

    const float bv = bias ? bias[n0 + r] : 0.0f;
    const int col = n0 + r;
#pragma unroll
    for (int j = 0; j < 8; ++j) {                 // C rows m0+g*8+j
        int row = m0 + g * 8 + j;
        float v = acc[j] + bv;
        if (DO_MISH) v = mishf(v);
        if (Ch) Ch[(size_t)row * N + col] = (_Float16)v;
        if (Cf) Cf[(size_t)row * N + col] = v;
    }
}

// ---------------- sequential GRU scan (persistent single workgroup) ---------
// h lives in LDS (f32). Thread j owns hidden unit j: three length-1024 dots
// against whh rows j / H+j / 2H+j (L2-resident, ~12.6 MB re-read per step).
// TDM double-buffers the next step's 12KB igates row into LDS behind compute.
__global__ void __launch_bounds__(1024)
k_gru_scan(const float* __restrict__ igates, const float* __restrict__ whh,
           const float* __restrict__ gru_bn, const int* __restrict__ start,
           const float* __restrict__ h0, _Float16* __restrict__ states_h,
           float* __restrict__ final_out) {
    __shared__ __align__(16) float h[HID];
#if HAVE_TDM
    __shared__ __align__(16) float ig_lds[2][3 * HID];
#endif
    const int j = threadIdx.x;
    h[j] = h0[j];
    const float bn = gru_bn[j];
    const float* wr = whh + (size_t)j * HID;
    const float* wz = whh + (size_t)(HID + j) * HID;
    const float* wn = whh + (size_t)(2 * HID + j) * HID;
#if HAVE_TDM
    if (j < 32) {                                  // wave 0 drives the TDM
        tdm_load_2d(lds_off(&ig_lds[0][0]), igates, 2, 3 * HID, 1, 3 * HID);
        tdm_wait();
    }
#endif
    __syncthreads();

    float hnew = h0[j];
    for (int t = 0; t < T_STEPS; ++t) {
#if HAVE_TDM
        if (t + 1 < T_STEPS && j < 32)             // prefetch next igates row
            tdm_load_2d(lds_off(&ig_lds[(t + 1) & 1][0]),
                        igates + (size_t)(t + 1) * (3 * HID), 2, 3 * HID, 1, 3 * HID);
        const float* ig = &ig_lds[t & 1][0];
#else
        const float* ig = igates + (size_t)t * (3 * HID);
#endif
        const float keep = (start[t] == 0) ? 1.0f : 0.0f;
        const float hin = h[j] * keep;
        __syncthreads();
        h[j] = hin;                                // h_in = h * (1 - start)
        __syncthreads();

        float ar = 0.f, az = 0.f, an = 0.f;
        for (int k = 0; k < HID; k += 4) {
            float4 hv  = *(const float4*)&h[k];
            float4 wrv = *(const float4*)(wr + k);
            float4 wzv = *(const float4*)(wz + k);
            float4 wnv = *(const float4*)(wn + k);
            ar += hv.x * wrv.x + hv.y * wrv.y + hv.z * wrv.z + hv.w * wrv.w;
            az += hv.x * wzv.x + hv.y * wzv.y + hv.z * wzv.z + hv.w * wzv.w;
            an += hv.x * wnv.x + hv.y * wnv.y + hv.z * wnv.z + hv.w * wnv.w;
        }

        float rg = 1.0f / (1.0f + __expf(-(ig[j] + ar)));
        float ug = 1.0f / (1.0f + __expf(-(ig[HID + j] + az)));
        float ng = tanhf(ig[2 * HID + j] + rg * (an + bn));
        hnew = ng + ug * (hin - ng);               // (1-u)*n + u*h_in
        states_h[(size_t)t * HID + j] = (_Float16)hnew;
        __syncthreads();
        h[j] = hnew;
#if HAVE_TDM
        if (t + 1 < T_STEPS) {
            if (j < 32) tdm_wait();                // next row landed
            __syncthreads();                       // publish before gate reads
        }
#endif
    }
    final_out[j] = hnew;
}

// ----------------------------- launcher -------------------------------------
extern "C" void kernel_launch(void* const* d_in, const int* in_sizes, int n_in,
                              void* d_out, int out_size, void* d_ws, size_t ws_size,
                              hipStream_t stream) {
    const float* x       = (const float*)d_in[0];
    const float* state   = (const float*)d_in[1];
    const int*   start   = (const int*)d_in[2];
    // d_in[3] = done (unused)
    const float* pre_wm  = (const float*)d_in[4];
    const float* pre_ws  = (const float*)d_in[5];
    const float* pre_bm  = (const float*)d_in[6];
    const float* pre_bs  = (const float*)d_in[7];
    const float* gru_wih = (const float*)d_in[8];
    const float* gru_whh = (const float*)d_in[9];
    const float* gru_b   = (const float*)d_in[10];
    const float* gru_bn  = (const float*)d_in[11];
    const float* p1_wm   = (const float*)d_in[12];
    const float* p1_ws   = (const float*)d_in[13];
    const float* p1_bm   = (const float*)d_in[14];
    const float* p1_bs   = (const float*)d_in[15];
    const float* p2_wm   = (const float*)d_in[16];
    const float* p2_ws   = (const float*)d_in[17];
    const float* p2_bm   = (const float*)d_in[18];
    const float* p2_bs   = (const float*)d_in[19];
    const float* p3_wm   = (const float*)d_in[20];
    const float* p3_ws   = (const float*)d_in[21];
    const float* p3_bm   = (const float*)d_in[22];
    const float* p3_bs   = (const float*)d_in[23];

    // workspace carve-out (256B aligned slices)
    char* wsb = (char*)d_ws;
    size_t off = 0;
    auto take = [&](size_t bytes) -> char* {
        char* p = wsb + off;
        off += (bytes + 255) & ~(size_t)255;
        return p;
    };
    _Float16* xh      = (_Float16*)take((size_t)T_STEPS * OBS * 2);
    _Float16* preWh   = (_Float16*)take((size_t)MLPD * OBS * 2);
    float*    preB    = (float*)   take((size_t)MLPD * 4);
    _Float16* zh      = (_Float16*)take((size_t)T_STEPS * MLPD * 2);
    _Float16* wihh    = (_Float16*)take((size_t)3 * HID * MLPD * 2);
    float*    ig      = (float*)   take((size_t)T_STEPS * 3 * HID * 4);
    _Float16* statesh = (_Float16*)take((size_t)T_STEPS * HID * 2);
    _Float16* p1Wh    = (_Float16*)take((size_t)MLPD * HID * 2);
    float*    p1B     = (float*)   take((size_t)MLPD * 4);
    _Float16* y1h     = (_Float16*)take((size_t)T_STEPS * MLPD * 2);
    _Float16* p2Wh    = (_Float16*)take((size_t)MLPD * MLPD * 2);
    float*    p2B     = (float*)   take((size_t)MLPD * 4);
    _Float16* y2h     = (_Float16*)take((size_t)T_STEPS * MLPD * 2);
    _Float16* p3Wh    = (_Float16*)take((size_t)ACT_DIM * MLPD * 2);
    float*    p3B     = (float*)   take((size_t)ACT_DIM * 4);

    const int TPB = 256;
    auto cdiv = [](int a, int b) { return (a + b - 1) / b; };

    // 1) materialize f16 activations / (noisy) weights / noisy biases
    k_noisy_f16<<<cdiv(T_STEPS * OBS, TPB), TPB, 0, stream>>>(x, nullptr, xh, T_STEPS * OBS, 0u);
    k_noisy_f16<<<cdiv(MLPD * OBS, TPB), TPB, 0, stream>>>(pre_wm, pre_ws, preWh, MLPD * OBS, 0x1001u);
    k_noisy_f32<<<cdiv(MLPD, TPB), TPB, 0, stream>>>(pre_bm, pre_bs, preB, MLPD, 0x1002u);
    k_noisy_f16<<<cdiv(3 * HID * MLPD, TPB), TPB, 0, stream>>>(gru_wih, nullptr, wihh, 3 * HID * MLPD, 0u);
    k_noisy_f16<<<cdiv(MLPD * HID, TPB), TPB, 0, stream>>>(p1_wm, p1_ws, p1Wh, MLPD * HID, 0x2001u);
    k_noisy_f32<<<cdiv(MLPD, TPB), TPB, 0, stream>>>(p1_bm, p1_bs, p1B, MLPD, 0x2002u);
    k_noisy_f16<<<cdiv(MLPD * MLPD, TPB), TPB, 0, stream>>>(p2_wm, p2_ws, p2Wh, MLPD * MLPD, 0x3001u);
    k_noisy_f32<<<cdiv(MLPD, TPB), TPB, 0, stream>>>(p2_bm, p2_bs, p2B, MLPD, 0x3002u);
    k_noisy_f16<<<cdiv(ACT_DIM * MLPD, TPB), TPB, 0, stream>>>(p3_wm, p3_ws, p3Wh, ACT_DIM * MLPD, 0x4001u);
    k_noisy_f32<<<cdiv(ACT_DIM, TPB), TPB, 0, stream>>>(p3_bm, p3_bs, p3B, ACT_DIM, 0x4002u);

    dim3 blk(256);
    // 2) z = mish(x @ preW^T + preB)           [2048,1024]
    k_wmma_gemm<1><<<dim3(MLPD / 16, T_STEPS / 128), blk, 0, stream>>>(
        xh, preWh, preB, zh, nullptr, T_STEPS, MLPD, OBS);
    // 3) igates = z @ wih^T + gru_b            [2048,3072] f32
    k_wmma_gemm<0><<<dim3(3 * HID / 16, T_STEPS / 128), blk, 0, stream>>>(
        zh, wihh, gru_b, nullptr, ig, T_STEPS, 3 * HID, MLPD);
    // 4) sequential GRU scan -> states (f16) + final_state (f32, tail of d_out)
    k_gru_scan<<<1, 1024, 0, stream>>>(
        ig, gru_whh, gru_bn, start, state, statesh,
        (float*)d_out + (size_t)T_STEPS * ACT_DIM);
    // 5) y1 = mish(states @ p1W^T + p1B)
    k_wmma_gemm<1><<<dim3(MLPD / 16, T_STEPS / 128), blk, 0, stream>>>(
        statesh, p1Wh, p1B, y1h, nullptr, T_STEPS, MLPD, HID);
    // 6) y2 = mish(y1 @ p2W^T + p2B)
    k_wmma_gemm<1><<<dim3(MLPD / 16, T_STEPS / 128), blk, 0, stream>>>(
        y1h, p2Wh, p2B, y2h, nullptr, T_STEPS, MLPD, MLPD);
    // 7) y = y2 @ p3W^T + p3B -> d_out[0 : T*ACT] (N=16 -> single WMMA N-tile)
    k_wmma_gemm<0><<<dim3(ACT_DIM / 16, T_STEPS / 128), blk, 0, stream>>>(
        y2h, p3Wh, p3B, nullptr, (float*)d_out, T_STEPS, ACT_DIM, MLPD);
}
